// SS2D_MixerLite_25872882991493
// MI455X (gfx1250) — matrix operand order repaired and checked by reading, not compiled, not run on hardware
//
#include <hip/hip_runtime.h>

typedef __bf16 bf16;
typedef __attribute__((ext_vector_type(16))) __bf16 v16bf;
typedef __attribute__((ext_vector_type(8)))  __bf16 v8bf;
typedef __attribute__((ext_vector_type(8)))  float  v8f;

#define B_  4
#define C_  256
#define H_  64
#define W_  64
#define L_  4096
#define M_  16384   /* B_ * L_ tokens */
#define HID 512
#define LDK 40      /* padded LDS K-stride (32 + 8 bf16) -> 80B rows, 16B aligned */

// ---------------- weight fp32 -> bf16 ----------------
__global__ __launch_bounds__(256) void k_f32_to_bf16(const float* __restrict__ in,
                                                     bf16* __restrict__ out, int n) {
  int i = blockIdx.x * 256 + threadIdx.x;
  if (i < n) out[i] = (bf16)in[i];
}

// ---------------- channel-LN of NCHW x -> bf16 [M, C] (NHWC tokens) ----------------
__global__ __launch_bounds__(256) void k_ln_x(const float* __restrict__ x,
                                              const float* __restrict__ w,
                                              const float* __restrict__ b,
                                              bf16* __restrict__ xn) {
  __shared__ float tile[32][C_ + 1];
  const int t  = threadIdx.x;
  const int p0 = blockIdx.x * 32;          // token base (32 pixels, same batch)
  const int bN = p0 >> 12;
  const int l0 = p0 & (L_ - 1);
  const int px = t & 31, cg = t >> 5;
  const float* xb = x + ((size_t)bN * C_) * L_ + l0;
  for (int j = 0; j < 32; ++j) {           // coalesced NCHW load, LDS transpose
    int c = j * 8 + cg;
    tile[px][c] = xb[(size_t)c * L_ + px];
  }
  __syncthreads();
  const int wid = t >> 5, lane = t & 31;
  for (int k = 0; k < 4; ++k) {
    int p = wid * 4 + k;
    float v[8], s = 0.f, ss = 0.f;
    for (int j = 0; j < 8; ++j) { v[j] = tile[p][lane + 32 * j]; s += v[j]; ss += v[j] * v[j]; }
    for (int m = 16; m; m >>= 1) { s += __shfl_xor(s, m, 32); ss += __shfl_xor(ss, m, 32); }
    float mu = s * (1.f / C_);
    float rs = rsqrtf(ss * (1.f / C_) - mu * mu + 1e-6f);
    bf16* o = xn + (size_t)(p0 + p) * C_;
    for (int j = 0; j < 8; ++j) {
      int c = lane + 32 * j;
      o[c] = (bf16)((v[j] - mu) * rs * w[c] + b[c]);
    }
  }
}

// ---------------- LN over last dim of row-major fp32 [M, C] -> bf16 ----------------
__global__ __launch_bounds__(256) void k_ln_row(const float* __restrict__ s,
                                                const float* __restrict__ w,
                                                const float* __restrict__ b,
                                                bf16* __restrict__ out) {
  const int t = threadIdx.x, wid = t >> 5, lane = t & 31;
  const size_t row = (size_t)blockIdx.x * 8 + wid;
  const float* r = s + row * C_;
  float v[8], sm = 0.f, ss = 0.f;
  for (int j = 0; j < 8; ++j) { v[j] = r[lane + 32 * j]; sm += v[j]; ss += v[j] * v[j]; }
  for (int m = 16; m; m >>= 1) { sm += __shfl_xor(sm, m, 32); ss += __shfl_xor(ss, m, 32); }
  float mu = sm * (1.f / C_);
  float rs = rsqrtf(ss * (1.f / C_) - mu * mu + 1e-6f);
  bf16* o = out + row * C_;
  for (int j = 0; j < 8; ++j) {
    int c = lane + 32 * j;
    o[c] = (bf16)((v[j] - mu) * rs * w[c] + b[c]);
  }
}

// ---------------- depthwise 3x3 on NHWC fp32 [M, C] ----------------
__global__ __launch_bounds__(256) void k_dwconv2d(const float* __restrict__ y,
                                                  const float* __restrict__ wk,
                                                  const float* __restrict__ bias,
                                                  float* __restrict__ out) {
  const size_t idx = (size_t)blockIdx.x * 256 + threadIdx.x;
  const int c = (int)(idx & (C_ - 1));
  const size_t tok = idx >> 8;
  const int l = (int)(tok & (L_ - 1));
  const int bN = (int)(tok >> 12);
  const int h = l >> 6, ww = l & 63;
  const float* wp = wk + c * 9;
  const float* yb = y + ((size_t)bN * L_) * C_;
  float acc = bias[c];
  #pragma unroll
  for (int kh = -1; kh <= 1; ++kh) {
    int hh = h + kh;
    if ((unsigned)hh >= (unsigned)H_) continue;
    #pragma unroll
    for (int kw = -1; kw <= 1; ++kw) {
      int wv = ww + kw;
      if ((unsigned)wv >= (unsigned)W_) continue;
      acc += wp[(kh + 1) * 3 + (kw + 1)] * yb[(size_t)(hh * W_ + wv) * C_ + c];
    }
  }
  out[tok * C_ + c] = acc;
}

// ---------------- SiLU gate: u = a * sigmoid(g), ag fp32 [M, 1024] ----------------
__global__ __launch_bounds__(256) void k_gate(const float* __restrict__ ag,
                                              float* __restrict__ u) {
  const size_t idx = (size_t)blockIdx.x * 256 + threadIdx.x;
  const int ch = (int)(idx & (HID - 1));
  const size_t tok = idx >> 9;
  float a = ag[tok * (2 * HID) + ch];
  float g = ag[tok * (2 * HID) + HID + ch];
  u[tok * HID + ch] = a * (1.f / (1.f + __expf(-g)));
}

// ---------------- 4-direction conv1d collapsed stencil, sum -> bf16 [M, HID] ----------------
__global__ __launch_bounds__(256) void k_convsum(const float* __restrict__ u,
                                                 const float* __restrict__ dw,
                                                 const float* __restrict__ dwb,
                                                 bf16* __restrict__ cs) {
  const size_t idx = (size_t)blockIdx.x * 256 + threadIdx.x;
  const int ch = (int)(idx & (HID - 1));
  const size_t tok = idx >> 9;
  const int l = (int)(tok & (L_ - 1));
  const size_t base = tok - (size_t)l;     // batch token base
  const float w0 = dw[ch * 3 + 0], w1 = dw[ch * 3 + 1], w2 = dw[ch * 3 + 2];
  float nb = 0.f;
  if (l >= 1)      nb += u[(base + l - 1) * HID + ch];              // row-scan fwd+rev
  if (l <= L_ - 2) nb += u[(base + l + 1) * HID + ch];
  const int h = l >> 6, wv = l & 63;
  const int p2 = wv * H_ + h;                                       // col-scan position
  if (p2 >= 1)      { int q = p2 - 1; nb += u[(base + (size_t)((q & 63) * W_ + (q >> 6))) * HID + ch]; }
  if (p2 <= L_ - 2) { int q = p2 + 1; nb += u[(base + (size_t)((q & 63) * W_ + (q >> 6))) * HID + ch]; }
  float r = (w0 + w2) * nb + 4.f * w1 * u[tok * HID + ch] + 4.f * dwb[ch];
  cs[tok * HID + ch] = (bf16)r;
}

// ---------------- bf16 WMMA GEMM: out = A[M,K] @ Wt[N,K]^T (+ fused epilogue) ----------------
// EPI 0: outF = acc + bias                         (fp32 [M,N])
// EPI 1: outH = bf16( res + 0.25*acc + bias )      (m = s + 0.25*sum_dirs)
// EPI 2: outF(NCHW) = nanclean(acc + bias) + res(NCHW x)
template <int EPI>
__global__ __launch_bounds__(256) void k_gemm(const bf16* __restrict__ A,
                                              const bf16* __restrict__ Wt,
                                              const float* __restrict__ bias,
                                              const float* __restrict__ res,
                                              float* __restrict__ outF,
                                              bf16* __restrict__ outH,
                                              int M, int N, int K) {
  __shared__ bf16 As[128 * LDK];
  __shared__ bf16 Bs[128 * LDK];
  const int t    = threadIdx.x;
  const int lane = t & 31;
  const int wid  = t >> 5;
  const int m0   = blockIdx.y * 128;
  const int n0   = blockIdx.x * 128;
  const int mt0  = (wid >> 1) * 32;        // wave tile: 32 (M) x 64 (N)
  const int nt0  = (wid & 1) * 64;
  const int lrow = lane & 15;
  const int kbA  = (lane & 16) ? 8 : 0;    // A frag: K in {kbA..kbA+7, kbA+16..kbA+23}
  const int kbB  = (lane & 16) ? 8 : 0;    // B frag: K in {kbB..kbB+7, kbB+16..kbB+23}
  const int ldr  = t >> 1;                 // staging row 0..127
  const int ldc  = (t & 1) * 16;           // staging col 0 / 16

  v8f acc[2][4] = {};

  const bf16* ga = A  + (size_t)(m0 + ldr) * K + ldc;
  const bf16* gb = Wt + (size_t)(n0 + ldr) * K + ldc;

  for (int k0 = 0; k0 < K; k0 += 32) {
    v8bf a0 = *(const v8bf*)(ga + k0);
    v8bf a1 = *(const v8bf*)(ga + k0 + 8);
    v8bf b0 = *(const v8bf*)(gb + k0);
    v8bf b1 = *(const v8bf*)(gb + k0 + 8);
    if (k0 + 32 < K) {
      __builtin_prefetch(ga + k0 + 32, 0, 1);
      __builtin_prefetch(gb + k0 + 32, 0, 1);
    }
    __syncthreads();
    *(v8bf*)&As[ldr * LDK + ldc]     = a0;
    *(v8bf*)&As[ldr * LDK + ldc + 8] = a1;
    *(v8bf*)&Bs[ldr * LDK + ldc]     = b0;
    *(v8bf*)&Bs[ldr * LDK + ldc + 8] = b1;
    __syncthreads();

    v16bf af[2], bfr[4];
    #pragma unroll
    for (int im = 0; im < 2; ++im) {
      const bf16* p = &As[(mt0 + im * 16 + lrow) * LDK];
      v8bf lo = *(const v8bf*)(p + kbA);
      v8bf hi = *(const v8bf*)(p + kbA + 16);
      af[im] = __builtin_shufflevector(lo, hi, 0,1,2,3,4,5,6,7,8,9,10,11,12,13,14,15);
    }
    #pragma unroll
    for (int jn = 0; jn < 4; ++jn) {
      const bf16* p = &Bs[(nt0 + jn * 16 + lrow) * LDK + kbB];
      v8bf lo = *(const v8bf*)(p);
      v8bf hi = *(const v8bf*)(p + 16);
      bfr[jn] = __builtin_shufflevector(lo, hi, 0,1,2,3,4,5,6,7,8,9,10,11,12,13,14,15);
    }
    #pragma unroll
    for (int im = 0; im < 2; ++im)
      #pragma unroll
      for (int jn = 0; jn < 4; ++jn)
        acc[im][jn] = __builtin_amdgcn_wmma_f32_16x16x32_bf16(
            false, af[im], false, bfr[jn], (short)0, acc[im][jn], false, false);
  }

  const int cn = lane & 15;
  const int mh = (lane >> 4) * 8;          // C layout: VGPR r -> M = mh + r, N = cn
  #pragma unroll
  for (int im = 0; im < 2; ++im) {
    #pragma unroll
    for (int jn = 0; jn < 4; ++jn) {
      const int gn = n0 + nt0 + jn * 16 + cn;
      const float bv = bias[gn];
      #pragma unroll
      for (int r = 0; r < 8; ++r) {
        const int gm = m0 + mt0 + im * 16 + mh + r;
        float v = acc[im][jn][r];
        if constexpr (EPI == 0) {
          outF[(size_t)gm * N + gn] = v + bv;
        } else if constexpr (EPI == 1) {
          float val = res[(size_t)gm * N + gn] + 0.25f * v + bv;
          outH[(size_t)gm * N + gn] = (bf16)val;
        } else {
          float val = v + bv;
          val = (val != val) ? 0.f : fminf(fmaxf(val, -10000.f), 10000.f);
          const int bN = gm >> 12;
          const int l  = gm & (L_ - 1);
          const size_t oidx = (((size_t)bN * C_ + gn) << 12) + l;  // NCHW scatter
          outF[oidx] = val + res[oidx];
        }
      }
    }
  }
}

extern "C" void kernel_launch(void* const* d_in, const int* in_sizes, int n_in,
                              void* d_out, int out_size, void* d_ws, size_t ws_size,
                              hipStream_t stream) {
  (void)in_sizes; (void)n_in; (void)out_size; (void)ws_size;
  const float* x         = (const float*)d_in[0];
  const float* ln2d_w    = (const float*)d_in[1];
  const float* ln2d_b    = (const float*)d_in[2];
  const float* inproj_w  = (const float*)d_in[3];
  const float* inproj_b  = (const float*)d_in[4];
  const float* dw2d_w    = (const float*)d_in[5];
  const float* dw2d_b    = (const float*)d_in[6];
  const float* ssm_ln_w  = (const float*)d_in[7];
  const float* ssm_ln_b  = (const float*)d_in[8];
  const float* ssm_in_w  = (const float*)d_in[9];
  const float* ssm_in_b  = (const float*)d_in[10];
  const float* ssm_dw_w  = (const float*)d_in[11];
  const float* ssm_dw_b  = (const float*)d_in[12];
  const float* ssm_out_w = (const float*)d_in[13];
  const float* ssm_out_b = (const float*)d_in[14];
  const float* outproj_w = (const float*)d_in[15];
  const float* outproj_b = (const float*)d_in[16];

  char* ws = (char*)d_ws;
  size_t off = 0;
  auto alloc = [&](size_t bytes) -> void* {
    void* p = ws + off;
    off += (bytes + 255) & ~(size_t)255;
    return p;
  };

  bf16*  bufA  = (bf16*) alloc((size_t)M_ * C_ * 2);        // xn -> sn -> m (reused)
  float* bufS  = (float*)alloc((size_t)M_ * C_ * 4);        // s tokens (residual)
  float* bufC  = (float*)alloc((size_t)M_ * 2 * HID * 4);   // y2 (front) -> ag
  float* bufU  = (float*)alloc((size_t)M_ * HID * 4);       // gated u
  bf16*  bufCS = (bf16*) alloc((size_t)M_ * HID * 2);       // conv-dir sum
  bf16*  wib   = (bf16*) alloc((size_t)C_ * C_ * 2);
  bf16*  wsib  = (bf16*) alloc((size_t)2 * HID * C_ * 2);
  bf16*  wsob  = (bf16*) alloc((size_t)C_ * HID * 2);
  bf16*  wob   = (bf16*) alloc((size_t)C_ * C_ * 2);
  float* y2 = bufC;

  dim3 blk(256);
  k_f32_to_bf16<<<(C_ * C_) / 256,       blk, 0, stream>>>(inproj_w,  wib,  C_ * C_);
  k_f32_to_bf16<<<(2 * HID * C_) / 256,  blk, 0, stream>>>(ssm_in_w,  wsib, 2 * HID * C_);
  k_f32_to_bf16<<<(C_ * HID) / 256,      blk, 0, stream>>>(ssm_out_w, wsob, C_ * HID);
  k_f32_to_bf16<<<(C_ * C_) / 256,       blk, 0, stream>>>(outproj_w, wob,  C_ * C_);

  k_ln_x<<<M_ / 32, blk, 0, stream>>>(x, ln2d_w, ln2d_b, bufA);
  k_gemm<0><<<dim3(C_ / 128, M_ / 128), blk, 0, stream>>>(bufA, wib, inproj_b, nullptr,
                                                          y2, nullptr, M_, C_, C_);
  k_dwconv2d<<<M_, blk, 0, stream>>>(y2, dw2d_w, dw2d_b, bufS);
  k_ln_row<<<M_ / 8, blk, 0, stream>>>(bufS, ssm_ln_w, ssm_ln_b, bufA);
  k_gemm<0><<<dim3((2 * HID) / 128, M_ / 128), blk, 0, stream>>>(bufA, wsib, ssm_in_b, nullptr,
                                                                 bufC, nullptr, M_, 2 * HID, C_);
  k_gate<<<((size_t)M_ * HID) / 256, blk, 0, stream>>>(bufC, bufU);
  k_convsum<<<((size_t)M_ * HID) / 256, blk, 0, stream>>>(bufU, ssm_dw_w, ssm_dw_b, bufCS);
  k_gemm<1><<<dim3(C_ / 128, M_ / 128), blk, 0, stream>>>(bufCS, wsob, ssm_out_b, bufS,
                                                          nullptr, bufA, M_, C_, HID);
  k_gemm<2><<<dim3(C_ / 128, M_ / 128), blk, 0, stream>>>(bufA, wob, outproj_b, x,
                                                          (float*)d_out, nullptr, M_, C_, C_);
}
